// ResNetPlusLSTM_62397284876573
// MI455X (gfx1250) — matrix operand order repaired and checked by reading, compiled, not run on hardware
//
#include <hip/hip_runtime.h>

// ---------------- problem constants (from reference) ----------------
constexpr int B_    = 4;
constexpr int S_    = 32;
constexpr int C_    = 128;
constexpr int H_    = 16;
constexpr int W_    = 32;
constexpr int HID_  = 128;
constexpr int DEPTH_= 2;

constexpr int HW_   = H_ * W_;          // 512 pixels per image
constexpr int NG_   = 4 * HID_;         // 512 gate channels
constexpr int CIN_  = C_ + HID_;        // 256 concat channels
constexpr int KTOT_ = CIN_ * 9;         // 2304 GEMM K
constexpr int MTOT_ = B_ * HW_;         // 2048 GEMM M (pixels)

// GEMM tiling
constexpr int MT_ = 64;                 // workgroup M tile (stays inside one batch image)
constexpr int NT_ = 256;                // workgroup N tile
constexpr int KT_ = 32;                 // K chunk (one bf16 WMMA K)
constexpr int KP_ = KT_ + 2;            // LDS row padding (1 DWORD -> bank spread)

typedef __attribute__((ext_vector_type(16))) __bf16 v16bf;
typedef __attribute__((ext_vector_type(8)))  float  v8f;
typedef unsigned int v4u __attribute__((ext_vector_type(4)));
typedef int v8i __attribute__((ext_vector_type(8)));
typedef int v4i __attribute__((ext_vector_type(4)));

union FragBF {
    v16bf    v;
    unsigned u[8];
};

#if __has_builtin(__builtin_amdgcn_tensor_load_to_lds) && __has_builtin(__builtin_amdgcn_s_wait_tensorcnt)
#define USE_TDM 1
#else
#define USE_TDM 0
#endif

// ---------------- weight repack: f32 -> bf16, n-major [D][N][K] (k contiguous) ----------------
// conv_w flat index (((d*NG+n)*CIN+ch)*9+t) == (d*NG+n)*KTOT + (ch*9+t), so this is a pure
// dtype-converting copy; K order (ch-major, 3x3 taps inner) matches the A-side k order.
__global__ void pack_weights_k(const float* __restrict__ cw, __bf16* __restrict__ wt)
{
    long idx   = (long)blockIdx.x * blockDim.x + threadIdx.x;
    long total = (long)DEPTH_ * NG_ * KTOT_;
    if (idx >= total) return;
    wt[idx] = (__bf16)cw[idx];
}

// ---------------- k -> (channel, dy, dx) lookup table (kills div/mod chains) ----------------
__global__ void fill_ktab_k(int* __restrict__ kt)
{
    int k = blockIdx.x * blockDim.x + threadIdx.x;
    if (k >= KTOT_) return;
    int ch = k / 9;
    int t  = k - ch * 9;
    kt[k] = ch | ((t / 3) << 16) | ((t % 3) << 20);
}

// ---------------- broadcast initial h/c states ----------------
__global__ void init_state_k(const float* __restrict__ ih, const float* __restrict__ ic,
                             float* __restrict__ hs, float* __restrict__ cs)
{
    int idx = blockIdx.x * blockDim.x + threadIdx.x;   // B*HID*HW = 262144
    if (idx >= B_ * HID_ * HW_) return;
    int u = (idx >> 9) & (HID_ - 1);
    hs[idx] = ih[u];
    cs[idx] = ic[u];
}

// ---------------- implicit-GEMM conv step: z = conv3x3(concat(x_t, h)) ----------------
// Block: 64(M) x 256(N); 8 wave32s as 2(M) x 4(N); wave tile 32x64 = 8 WMMAs / K-chunk.
__global__ __launch_bounds__(256) void conv_step_k(
    const float*  __restrict__ xin,  long xinBStride,
    const float*  __restrict__ hstate,
    const __bf16* __restrict__ wt,      // [NG][KTOT] n-major for this depth
    const int*    __restrict__ ktab,
    float*        __restrict__ zbuf)
{
    __shared__ __bf16 As [MT_][KP_];     // [m][k]
    __shared__ __bf16 BsT[NT_][KP_];     // [n][k]

    const int tid  = threadIdx.x;
    const int lane = tid & 31;
    const int wave = tid >> 5;           // 0..7 (wave32)
    const int wm   = wave & 1;           // M half (32 rows)
    const int wn   = wave >> 1;          // N quarter (64 cols)
    const int mtile = blockIdx.x * MT_;
    const int ntile = blockIdx.y * NT_;

    const int mrow = lane & 15;
    const int half = lane >> 4;

    const int bimg  = mtile >> 9;        // whole tile lives in one batch image (64 | 512)
    const int pbase = mtile & 511;

    v8f acc[2][4] = {};

#if USE_TDM
    // TDM descriptor pieces that are loop-invariant
    const unsigned ldsB = (unsigned)(unsigned long long)(void*)&BsT[0][0];
#endif

    for (int k0 = 0; k0 < KTOT_; k0 += KT_) {
        // ---- stage A: 64 pixels x 32 k; each thread keeps one fixed k of the chunk ----
        {
            int kk = tid & 31;
            int e  = ktab[k0 + kk];
            int ch = e & 0xFFFF;
            int dy = ((e >> 16) & 3) - 1;
            int dx = ((e >> 20) & 3) - 1;
            const float* src = (ch < C_)
                ? xin + (long)bimg * xinBStride + (long)ch * HW_
                : hstate + ((long)bimg * HID_ + (ch - C_)) * HW_;
            int mlb = tid >> 5;          // 0..7
            #pragma unroll
            for (int t2 = 0; t2 < MT_ / 8; ++t2) {
                int ml = mlb + t2 * 8;
                int p  = pbase + ml;
                int y  = (p >> 5) + dy;
                int x  = (p & 31) + dx;
                float v = 0.f;
                if ((unsigned)y < (unsigned)H_ && (unsigned)x < (unsigned)W_)
                    v = src[(y << 5) + x];
                As[ml][kk] = (__bf16)v;
            }
        }

        // ---- stage B: 256 n x 32 k tile of the n-major weights ----
#if USE_TDM
        if (wave == 0) {
            unsigned long long ga =
                (unsigned long long)(wt + (long)ntile * KTOT_ + k0);
            v4u g0 = {};
            g0[0] = 1u;                                    // count=1, user descriptor
            g0[1] = ldsB;                                  // lds_addr
            g0[2] = (unsigned)ga;                          // global_addr[31:0]
            g0[3] = (unsigned)((ga >> 32) & 0x1FFFFFFu) | (2u << 30);  // addr hi | type=2
            v8i g1 = {};
            g1[0] = (1 << 16) | (1 << 20) | (3 << 22);     // data_size=2B, pad_en, every 16 DW
            g1[1] = (KTOT_ & 0xFFFF) << 16;                // tensor_dim0 lo
            g1[2] = (NG_ & 0xFFFF) << 16;                  // dim0 hi=0 | tensor_dim1 lo
            g1[3] = (KT_ << 16);                           // dim1 hi=0 | tile_dim0=32
            g1[4] = NT_;                                   // tile_dim1=256
            g1[5] = KTOT_;                                 // tensor_dim0_stride lo32
            g1[6] = 0;
            g1[7] = 0;
            v4i z4 = {};
#if __has_include(<hip/amd_detail/amd_gfx1250_TDM.h>)
            v8i z8 = {};
            __builtin_amdgcn_tensor_load_to_lds(g0, g1, z4, z4, z8, 0);
#else
            __builtin_amdgcn_tensor_load_to_lds(g0, g1, z4, z4, 0);
#endif
            __builtin_amdgcn_s_wait_tensorcnt((short)0);
        }
#else
        {
            const unsigned* src =
                (const unsigned*)(wt + (long)(ntile + tid) * KTOT_ + k0);
            unsigned* dst = (unsigned*)&BsT[tid][0];
            #pragma unroll
            for (int i = 0; i < 16; ++i) dst[i] = src[i];
        }
#endif
        __syncthreads();

        // ---- A fragments (16-bit A 16x32 layout: lane half picks K offset 8 per 16-group) ----
        FragBF fa0, fa1;
        {
            const int mb = wm * 32 + mrow;
            #pragma unroll
            for (int q = 0; q < 4; ++q) {
                fa0.u[q]     = *(const unsigned*)&As[mb][half * 8 + 2 * q];
                fa0.u[4 + q] = *(const unsigned*)&As[mb][16 + half * 8 + 2 * q];
                fa1.u[q]     = *(const unsigned*)&As[mb + 16][half * 8 + 2 * q];
                fa1.u[4 + q] = *(const unsigned*)&As[mb + 16][16 + half * 8 + 2 * q];
            }
        }
        // ---- 4 B fragments, each reused by both A fragments: 8 WMMAs ----
        #pragma unroll
        for (int j = 0; j < 4; ++j) {
            FragBF fb;
            const int nn = wn * 64 + j * 16 + mrow;
            #pragma unroll
            for (int q = 0; q < 8; ++q)
                fb.u[q] = *(const unsigned*)&BsT[nn][(q >> 2) * 16 + half * 8 + 2 * (q & 3)];
            acc[0][j] = __builtin_amdgcn_wmma_f32_16x16x32_bf16(
                false, fa0.v, false, fb.v, (short)0, acc[0][j], false, false);
            acc[1][j] = __builtin_amdgcn_wmma_f32_16x16x32_bf16(
                false, fa1.v, false, fb.v, (short)0, acc[1][j], false, false);
        }
        __syncthreads();
    }

    // ---- epilogue: C/D layout (lane half -> M offset 8, VGPR r -> M row) ----
    #pragma unroll
    for (int a = 0; a < 2; ++a) {
        #pragma unroll
        for (int j = 0; j < 4; ++j) {
            int n = ntile + wn * 64 + j * 16 + mrow;
            #pragma unroll
            for (int r = 0; r < 8; ++r) {
                int mg = mtile + wm * 32 + a * 16 + r + half * 8;
                zbuf[(long)mg * NG_ + n] = acc[a][j][r];
            }
        }
    }
}

// ---------------- pointwise LSTM gate update ----------------
__global__ void gate_step_k(const float* __restrict__ zbuf, const float* __restrict__ bias,
                            float* __restrict__ cstate, float* __restrict__ hstate,
                            float* __restrict__ hseq_s)
{
    int idx = blockIdx.x * blockDim.x + threadIdx.x;   // B*HID*HW
    if (idx >= B_ * HID_ * HW_) return;
    int p = idx & 511;
    int u = (idx >> 9) & (HID_ - 1);
    int b = idx >> 16;
    long m = (long)b * HW_ + p;
    float zi = zbuf[m * NG_ + u          ] + bias[u          ];
    float zf = zbuf[m * NG_ + u + HID_   ] + bias[u + HID_   ];
    float zg = zbuf[m * NG_ + u + 2*HID_ ] + bias[u + 2*HID_ ];
    float zo = zbuf[m * NG_ + u + 3*HID_ ] + bias[u + 3*HID_ ];
    float si = 1.f / (1.f + __expf(-zi));
    float sf = 1.f / (1.f + __expf(-zf));
    float so = 1.f / (1.f + __expf(-zo));
    float g  = tanhf(zg);
    float c  = sf * cstate[idx] + si * g;
    float h  = so * tanhf(c);
    cstate[idx] = c;
    hstate[idx] = h;
    hseq_s[idx] = h;
}

// ---------------- global average pool over H,W ----------------
__global__ void pool_k(const float* __restrict__ hseq, float* __restrict__ pooled)
{
    int bs = blockIdx.x;          // (b,s)
    int b  = bs / S_;
    int s  = bs % S_;
    int u  = threadIdx.x;         // 0..HID-1
    const float* src = hseq + (((long)s * B_ + b) * HID_ + u) * HW_;
    float sum = 0.f;
    for (int p = 0; p < HW_; ++p) sum += src[p];
    pooled[(long)bs * HID_ + u] = sum * (1.f / (float)HW_);
}

// ---------------- FC head: relu(pooled @ fc_w.T + fc_b) -> offset/angle ----------------
__global__ void head_k(const float* __restrict__ pooled,
                       const float* __restrict__ fcw, const float* __restrict__ fcb,
                       const float* __restrict__ fow, const float* __restrict__ fob,
                       const float* __restrict__ faw, const float* __restrict__ fab,
                       float* __restrict__ out)
{
    __shared__ float po[HID_];
    __shared__ float ro[HID_];
    __shared__ float ra[HID_];
    int bs = blockIdx.x;
    int j  = threadIdx.x;
    po[j] = pooled[(long)bs * HID_ + j];
    __syncthreads();
    float f = fcb[j];
    for (int h = 0; h < HID_; ++h) f += po[h] * fcw[j * HID_ + h];
    f = fmaxf(f, 0.f);
    ro[j] = f * fow[j];
    ra[j] = f * faw[j];
    __syncthreads();
    for (int st = HID_ / 2; st > 0; st >>= 1) {
        if (j < st) { ro[j] += ro[j + st]; ra[j] += ra[j + st]; }
        __syncthreads();
    }
    if (j == 0) {
        out[bs]         = ro[0] + fob[0];   // offset [B,S,1] flat
        out[B_*S_ + bs] = ra[0] + fab[0];   // angle  [B,S,1] flat
    }
}

// ---------------- host orchestration ----------------
extern "C" void kernel_launch(void* const* d_in, const int* in_sizes, int n_in,
                              void* d_out, int out_size, void* d_ws, size_t ws_size,
                              hipStream_t stream)
{
    (void)in_sizes; (void)n_in; (void)out_size; (void)ws_size;
    const float* x      = (const float*)d_in[0];
    const float* conv_w = (const float*)d_in[1];
    const float* conv_b = (const float*)d_in[2];
    const float* init_h = (const float*)d_in[3];
    const float* init_c = (const float*)d_in[4];
    const float* fc_w   = (const float*)d_in[5];
    const float* fc_b   = (const float*)d_in[6];
    const float* fco_w  = (const float*)d_in[7];
    const float* fco_b  = (const float*)d_in[8];
    const float* fca_w  = (const float*)d_in[9];
    const float* fca_b  = (const float*)d_in[10];

    char* ws = (char*)d_ws;
    size_t off = 0;
    auto alloc = [&](size_t bytes) -> void* {
        void* p = ws + off;
        off += (bytes + 255) & ~(size_t)255;
        return p;
    };
    __bf16* wt     = (__bf16*)alloc((size_t)DEPTH_ * NG_ * KTOT_ * sizeof(__bf16)); // 4.7 MB
    int*    ktab   = (int*)   alloc((size_t)KTOT_ * sizeof(int));                   // 9 KB
    float*  hstate = (float*) alloc((size_t)B_ * HID_ * HW_ * sizeof(float));       // 1 MB
    float*  cstate = (float*) alloc((size_t)B_ * HID_ * HW_ * sizeof(float));       // 1 MB
    float*  zbuf   = (float*) alloc((size_t)MTOT_ * NG_ * sizeof(float));           // 4 MB
    float*  hseq   = (float*) alloc((size_t)S_ * B_ * HID_ * HW_ * sizeof(float));  // 32 MB
    float*  pooled = (float*) alloc((size_t)B_ * S_ * HID_ * sizeof(float));        // 64 KB

    // 1) weight repack + k-decomposition table
    {
        long total = (long)DEPTH_ * NG_ * KTOT_;
        pack_weights_k<<<(unsigned)((total + 255) / 256), 256, 0, stream>>>(conv_w, wt);
        fill_ktab_k<<<(KTOT_ + 255) / 256, 256, 0, stream>>>(ktab);
    }

    const int  stN = (B_ * HID_ * HW_ + 255) / 256;
    const dim3 gGemm(MTOT_ / MT_, NG_ / NT_);   // (32, 2)

    // 2) stacked ConvLSTM scan
    for (int d = 0; d < DEPTH_; ++d) {
        init_state_k<<<stN, 256, 0, stream>>>(init_h + d * HID_, init_c + d * HID_,
                                              hstate, cstate);
        const __bf16* wd = wt + (long)d * NG_ * KTOT_;
        for (int s = 0; s < S_; ++s) {
            const float* xin;
            long bstride;
            if (d == 0) {
                xin = x + (long)s * C_ * HW_;            // x[b][s][c][p]
                bstride = (long)S_ * C_ * HW_;
            } else {
                xin = hseq + (long)s * B_ * HID_ * HW_;  // previous depth's h at step s
                bstride = (long)HID_ * HW_;
            }
            conv_step_k<<<gGemm, 256, 0, stream>>>(xin, bstride, hstate, wd, ktab, zbuf);
            gate_step_k<<<stN, 256, 0, stream>>>(zbuf, conv_b + d * NG_, cstate, hstate,
                                                 hseq + (long)s * B_ * HID_ * HW_);
        }
    }

    // 3) pool + heads
    pool_k<<<B_ * S_, HID_, 0, stream>>>(hseq, pooled);
    head_k<<<B_ * S_, HID_, 0, stream>>>(pooled, fc_w, fc_b, fco_w, fco_b,
                                         fca_w, fca_b, (float*)d_out);
}
